// MultiHeadAttentionBlock_19481971654935
// MI455X (gfx1250) — matrix operand order repaired and checked by reading, not compiled, not run on hardware
//
#include <hip/hip_runtime.h>
#include <math.h>

#define BATCH  4
#define SEQ    2048
#define DMODEL 128
#define NHEAD  4
#define DHEAD  32
#define KT     64          // keys per attention iteration

typedef unsigned short u16;
typedef __bf16 bf16;
typedef bf16  v16bf __attribute__((ext_vector_type(16)));
typedef u16   v16u  __attribute__((ext_vector_type(16)));
typedef float v8f   __attribute__((ext_vector_type(8)));

// fp32 -> bf16 bits, round-to-nearest-even
__device__ inline u16 f2bf(float f) {
  unsigned u = __builtin_bit_cast(unsigned, f);
  unsigned r = 0x7FFFu + ((u >> 16) & 1u);
  return (u16)((u + r) >> 16);
}

// ---- WMMA 16x16x32 bf16 fragment loaders (contiguous k order, same for A and B) ----
// A: 16x32 (MxK). Lane L: row M = L%16. hi=L>>4 selects K-half:
//   j=0..15 -> K = hi*16 + j
__device__ inline v16bf load_a(const u16* p, int stride) {
  int lane = threadIdx.x & 31;
  int rm = lane & 15, hi = lane >> 4;
  const u16* q = p + rm * stride + (hi << 4);
  v16u a;
#pragma unroll
  for (int j = 0; j < 16; ++j)
    a[j] = q[j];
  return __builtin_bit_cast(v16bf, a);
}

// B: 32x16 (KxN), source stored "n-major": S[n][k] (already transposed).
// Lane L: col N = L%16; lanes 0-15 hold K=0..15, lanes 16-31 hold K=16..31.
__device__ inline v16bf load_b(const u16* p, int stride) {
  int lane = threadIdx.x & 31;
  int nm = lane & 15, hi = lane >> 4;
  const u16* q = p + nm * stride + (hi << 4);
  v16u b;
#pragma unroll
  for (int j = 0; j < 16; ++j) b[j] = q[j];
  return __builtin_bit_cast(v16bf, b);
}

__device__ inline v8f wmma_bf16(v16bf a, v16bf b, v8f c) {
  return __builtin_amdgcn_wmma_f32_16x16x32_bf16(false, a, false, b, (short)0, c,
                                                 false, false);
}

// async 16B global -> LDS copy (GLOBAL_LOAD_ASYNC_TO_LDS_B128, ASYNCcnt-tracked)
__device__ inline void async_copy16(void* lds_dst, const void* gsrc) {
  unsigned ldsoff = (unsigned)(size_t)lds_dst;   // addrspace(3) -> LDS byte offset
  asm volatile("global_load_async_to_lds_b128 %0, %1, off"
               :: "v"(ldsoff), "v"(gsrc) : "memory");
}
__device__ inline void wait_async0() {
  asm volatile("s_wait_asynccnt 0" ::: "memory");
}

// ---------------------------------------------------------------------------
// Generic 8192x128 @ 128x128^T projection, specialized per MODE:
//  0: q-proj  -> outF = q fp32 [B,L,D], outH = qh bf16 [H][B*L][32] PRESCALED by 1/sqrt(128)
//  1: k-proj  -> outH = kh bf16 [H][B*L][32]
//  2: v-proj  -> outH = vT bf16 [H][B][32][L]   (pre-transposed for A@V B-operand)
//  3: out FFN -> outF = resid + relu(X@W^T + bias)
// ---------------------------------------------------------------------------
template <int MODE>
__global__ __launch_bounds__(256)
void proj_kernel(const float* __restrict__ X, const float* __restrict__ W,
                 const float* __restrict__ bias,
                 float* __restrict__ outF, u16* __restrict__ outH,
                 const float* __restrict__ resid) {
  __shared__ u16 Xs[64][136];    // padded: 4-bank skew per row
  __shared__ u16 Ws[128][136];

  const int m0  = blockIdx.x * 64;
  const int tid = threadIdx.x;

  for (int i = tid; i < 64 * 128; i += 256) {
    int r = i >> 7, c = i & 127;
    Xs[r][c] = f2bf(X[(m0 + r) * DMODEL + c]);
  }
  for (int i = tid; i < 128 * 128; i += 256) {
    int r = i >> 7, c = i & 127;
    Ws[r][c] = f2bf(W[i]);
  }
  __syncthreads();

  const int wave = tid >> 5, lane = tid & 31;
  const int mrow0 = (wave & 3) * 16;
  const int ncol0 = (wave >> 2) * 64;
  const int rm = lane & 15, hi = lane >> 4;

  v16bf a[4];
#pragma unroll
  for (int kb = 0; kb < 4; ++kb) a[kb] = load_a(&Xs[mrow0][kb * 32], 136);

#pragma unroll
  for (int nb = 0; nb < 4; ++nb) {
    int n0 = ncol0 + nb * 16;
    v8f acc = {};
#pragma unroll
    for (int kb = 0; kb < 4; ++kb) {
      v16bf bf = load_b(&Ws[n0][kb * 32], 136);   // (k,n) -> W[n][k] == X@W^T
      acc = wmma_bf16(a[kb], bf, acc);
    }
    int col = n0 + rm;
    float bv = bias[col];
#pragma unroll
    for (int r = 0; r < 8; ++r) {
      int mg = m0 + mrow0 + r + (hi << 3);        // C layout: M = r + 8*hi
      float val = acc[r] + bv;
      if constexpr (MODE == 0) {
        outF[mg * DMODEL + col] = val;
        int h = col >> 5, d = col & 31;
        // fold score scale 1/sqrt(128) into q
        outH[(h * (BATCH * SEQ) + mg) * DHEAD + d] =
            f2bf(val * 0.0883883476483184f);
      } else if constexpr (MODE == 1) {
        int h = col >> 5, d = col & 31;
        outH[(h * (BATCH * SEQ) + mg) * DHEAD + d] = f2bf(val);
      } else if constexpr (MODE == 2) {
        int h = col >> 5, d = col & 31;
        int b = mg >> 11, l = mg & (SEQ - 1);
        outH[((h * BATCH + b) * DHEAD + d) * SEQ + l] = f2bf(val);
      } else {
        float x = resid[mg * DMODEL + col];
        outF[mg * DMODEL + col] = x + fmaxf(val, 0.0f);
      }
    }
  }
}

// ---------------------------------------------------------------------------
// Flash attention: block = (batch b, 16-query tile), 4 waves = 4 heads.
// Mask tile is head-independent -> staged once per block via async global->LDS
// (double-buffered; next chunk issued at top of iteration, drained at the end).
// Per wave per 64-key chunk: 4 score WMMAs, online softmax (deferred lane-local
// row sums), P via padded LDS -> A-layout, 4 A@V WMMAs.
// ---------------------------------------------------------------------------
__global__ __launch_bounds__(128)
void attn_kernel(const u16* __restrict__ qh, const u16* __restrict__ kh,
                 const u16* __restrict__ vT, const int* __restrict__ mask,
                 const float* __restrict__ qf, float* __restrict__ omid) {
  __shared__ int Msk[2][16][68];   // double-buffered mask tile, bank-skewed
  __shared__ u16 Pt[4][16][66];    // per-wave P tile, odd-bank skew

  const int qt  = blockIdx.x;
  const int b   = blockIdx.y;
  const int tid = threadIdx.x;
  const int h    = tid >> 5;       // wave == head
  const int lane = tid & 31;
  const int rm = lane & 15, hi = lane >> 4;
  const int q0 = qt * 16;

  const u16* qbase = qh + (h * BATCH + b) * SEQ * DHEAD + q0 * DHEAD;
  const u16* kbase = kh + (h * BATCH + b) * SEQ * DHEAD;
  const u16* vbase = vT + ((size_t)(h * BATCH + b) * DHEAD) * SEQ;
  const int* mbase = mask + ((size_t)b * SEQ + q0) * SEQ;

  // async mask chunk loader: 16 rows x 64 keys, 16B per lane per issue
  auto issue_mask = [&](int kc, int buf) {
    int k0 = kc * KT;
#pragma unroll
    for (int j = 0; j < 2; ++j) {
      int f = tid * 2 + j;                 // 0..255
      int row = f >> 4, seg = f & 15;
      async_copy16(&Msk[buf][row][seg * 4],
                   mbase + (size_t)row * SEQ + k0 + seg * 4);
    }
  };

  v16bf aq = load_a(qbase, DHEAD);

  v8f acc0 = {}, acc1 = {};
  float rmax[8], rsum[8];
#pragma unroll
  for (int r = 0; r < 8; ++r) { rmax[r] = -1e30f; rsum[r] = 0.0f; }
  const v8f zero = {};

  issue_mask(0, 0);
  wait_async0();
  __syncthreads();

  for (int kc = 0; kc < SEQ / KT; ++kc) {
    int k0 = kc * KT;
    const int cur = kc & 1;

    // overlap: async-stage next mask chunk + prefetch next K/V behind compute
    if (kc + 1 < SEQ / KT) {
      issue_mask(kc + 1, cur ^ 1);
      __builtin_prefetch(kbase + (size_t)(k0 + KT) * DHEAD, 0, 1);
      __builtin_prefetch(vbase + k0 + KT, 0, 1);
      __builtin_prefetch(vbase + 16 * SEQ + k0 + KT, 0, 1);
    }

    v8f s[4];
#pragma unroll
    for (int t = 0; t < 4; ++t) {
      v16bf bk = load_b(kbase + (size_t)(k0 + 16 * t) * DHEAD, DHEAD);
      s[t] = wmma_bf16(aq, bk, zero);      // q prescaled -> raw scores
    }

#pragma unroll
    for (int r = 0; r < 8; ++r) {
      const int* mr = &Msk[cur][r + (hi << 3)][rm];
      float x0 = (mr[0]  != 0) ? s[0][r] : -1e10f;
      float x1 = (mr[16] != 0) ? s[1][r] : -1e10f;
      float x2 = (mr[32] != 0) ? s[2][r] : -1e10f;
      float x3 = (mr[48] != 0) ? s[3][r] : -1e10f;
      float mx = fmaxf(fmaxf(fmaxf(x0, x1), fmaxf(x2, x3)), rmax[r]);
#pragma unroll
      for (int off = 8; off >= 1; off >>= 1)
        mx = fmaxf(mx, __shfl_xor(mx, off, 32));   // row-uniform max
      float corr = __expf(rmax[r] - mx);
      rmax[r] = mx;
      float p0 = __expf(x0 - mx), p1 = __expf(x1 - mx);
      float p2 = __expf(x2 - mx), p3 = __expf(x3 - mx);
      rsum[r] = rsum[r] * corr + ((p0 + p1) + (p2 + p3)); // lane-local partial
      acc0[r] *= corr;
      acc1[r] *= corr;
      u16* pr = &Pt[h][r + (hi << 3)][rm];
      pr[0]  = f2bf(p0);  pr[16] = f2bf(p1);
      pr[32] = f2bf(p2);  pr[48] = f2bf(p3);
    }
    asm volatile("s_wait_dscnt 0" ::: "memory");   // same-wave LDS RAW on Pt
    v16bf ap0  = load_a(&Pt[h][0][0],  66);        // keys k0..+31 in A-layout
    v16bf ap1  = load_a(&Pt[h][0][32], 66);        // keys +32..+63
    v16bf bv00 = load_b(vbase + k0, SEQ);               // k-chunk0, dh 0..15
    v16bf bv01 = load_b(vbase + 16 * SEQ + k0, SEQ);    // k-chunk0, dh 16..31
    v16bf bv10 = load_b(vbase + k0 + 32, SEQ);          // k-chunk1, dh 0..15
    v16bf bv11 = load_b(vbase + 16 * SEQ + k0 + 32, SEQ);
    acc0 = wmma_bf16(ap0, bv00, acc0);
    acc1 = wmma_bf16(ap0, bv01, acc1);
    acc0 = wmma_bf16(ap1, bv10, acc0);
    acc1 = wmma_bf16(ap1, bv11, acc1);

    wait_async0();                         // drain this wave's async stores
    __syncthreads();                       // all waves' chunks visible
  }

  // deferred cross-lane row-sum reduction (16-lane groups)
#pragma unroll
  for (int r = 0; r < 8; ++r) {
#pragma unroll
    for (int off = 8; off >= 1; off >>= 1)
      rsum[r] += __shfl_xor(rsum[r], off, 32);
  }

#pragma unroll
  for (int r = 0; r < 8; ++r) {
    int qrow = q0 + r + (hi << 3);
    float inv = 1.0f / rsum[r];
    int idx = ((size_t)b * SEQ + qrow) * DMODEL + h * DHEAD + rm;
    omid[idx]      = qf[idx]      + acc0[r] * inv;  // residual: oh = qh + A@vh
    omid[idx + 16] = qf[idx + 16] + acc1[r] * inv;
  }
}

// ---------------------------------------------------------------------------
extern "C" void kernel_launch(void* const* d_in, const int* in_sizes, int n_in,
                              void* d_out, int out_size, void* d_ws, size_t ws_size,
                              hipStream_t stream) {
  (void)in_sizes; (void)n_in; (void)out_size; (void)ws_size;
  const float* Q    = (const float*)d_in[0];
  const float* K    = (const float*)d_in[1];
  const int*   mask = (const int*)d_in[2];
  const float* Wq   = (const float*)d_in[3];
  const float* bq   = (const float*)d_in[4];
  const float* Wk   = (const float*)d_in[5];
  const float* bk   = (const float*)d_in[6];
  const float* Wv   = (const float*)d_in[7];
  const float* bv   = (const float*)d_in[8];
  const float* Wo   = (const float*)d_in[9];
  const float* bo   = (const float*)d_in[10];

  char* ws = (char*)d_ws;
  float* q_f32 = (float*)ws;  ws += (size_t)BATCH * SEQ * DMODEL * 4;
  float* o_mid = (float*)ws;  ws += (size_t)BATCH * SEQ * DMODEL * 4;
  u16*   qh    = (u16*)ws;    ws += (size_t)NHEAD * BATCH * SEQ * DHEAD * 2;
  u16*   kh    = (u16*)ws;    ws += (size_t)NHEAD * BATCH * SEQ * DHEAD * 2;
  u16*   vT    = (u16*)ws;

  dim3 gb(BATCH * SEQ / 64), tb(256);
  proj_kernel<0><<<gb, tb, 0, stream>>>(Q, Wq, bq, q_f32, qh, nullptr);
  proj_kernel<1><<<gb, tb, 0, stream>>>(K, Wk, bk, nullptr, kh, nullptr);
  proj_kernel<2><<<gb, tb, 0, stream>>>(K, Wv, bv, nullptr, vT, nullptr);

  attn_kernel<<<dim3(SEQ / 16, BATCH), dim3(128), 0, stream>>>(qh, kh, vT, mask,
                                                               q_f32, o_mid);

  proj_kernel<3><<<gb, tb, 0, stream>>>(o_mid, Wo, bo, (float*)d_out, nullptr, o_mid);
}
